// MoEGate_43963285242657
// MI455X (gfx1250) — matrix operand order repaired and checked by reading, not compiled, not run on hardware
//
#include <hip/hip_runtime.h>
#include <hip/hip_bf16.h>

// ---------------------------------------------------------------------------
// MoE gate for MI455X (gfx1250, wave32).
//   scores = sigmoid(X @ W^T)  via split-precision bf16 WMMA (bf16x3):
//   x = hi + lo;  acc += hi*hi + lo*hi + hi*lo  (f32 accumulator)
//   -> ~2^-18 relative logit error, preserves fp32 top-k ordering.
// W (4 MB) is pre-split once into an interleaved bf16 hi/lo device global
// ([32 hi][32 lo] per K-block -> one base + immediate offsets per fragment).
// Hot loop: direct B fragment loads; per n-tile the two b_hi uses are
// ADJACENT (hi*hi then lo*hi) so b_hi live ranges stay short -- this stops
// the allocator from batching 8 b_hi fragments and full-draining loadcnt on
// every b_lo reload. A raw data single-buffered one K-step ahead.
// ---------------------------------------------------------------------------

typedef __attribute__((ext_vector_type(16))) __bf16 v16bf;
typedef __attribute__((ext_vector_type(8)))  __bf16 v8bf;
typedef __attribute__((ext_vector_type(8)))  float  v8f;

constexpr int   kT = 16384;   // tokens
constexpr int   kH = 4096;    // hidden
constexpr int   kE = 256;     // experts
constexpr int   kTopK = 8;
constexpr float kScale = 2.5f;

constexpr int kTokPerBlock = 64;   // 4 waves: 2 token groups x 2 expert halves
constexpr int kThreads     = 128;
constexpr int kSteps       = kH / 32;   // 128 K-steps

// Pre-split W: per 32-element K-block, 32 bf16 hi then 32 bf16 lo (128 B).
__device__ __attribute__((aligned(32))) __bf16 g_w_split[(size_t)kE * kH * 2];

__device__ __forceinline__ v8f wmma_bf16(v16bf a, v16bf b, v8f c) {
  // (neg_a, A, neg_b, B, c_mod, C, reuse_a, reuse_b)
  return __builtin_amdgcn_wmma_f32_16x16x32_bf16(false, a, false, b,
                                                 (short)0, c, false, false);
}

// Convert 8 consecutive f32 (two float4) into bf16 hi + residual lo,
// writing fragment elements [off .. off+7].
__device__ __forceinline__ void cvt_split8(float4 a, float4 b,
                                           v16bf& hi, v16bf& lo, int off) {
  float v[8] = {a.x, a.y, a.z, a.w, b.x, b.y, b.z, b.w};
#pragma unroll
  for (int i = 0; i < 8; ++i) {
    __bf16 h = (__bf16)v[i];               // RNE f32 -> bf16
    hi[off + i] = h;
    lo[off + i] = (__bf16)(v[i] - (float)h);
  }
}

__device__ __forceinline__ float sigmoidf_fast(float v) {
  return 1.0f / (1.0f + __expf(-v));
}

// ------------------------- prep: split W into bf16 hi/lo --------------------
// One thread per 8 elements. E*H = 1M elements -> 512 blocks x 256 threads.
// Element i lands at block (i/32): hi at blk*64 + (i%32), lo at +32.
__global__ void split_w_kernel(const float* __restrict__ w) {
  const size_t i = ((size_t)blockIdx.x * blockDim.x + threadIdx.x) * 8;
  float4 a = *(const float4*)(w + i);
  float4 b = *(const float4*)(w + i + 4);
  float v[8] = {a.x, a.y, a.z, a.w, b.x, b.y, b.z, b.w};
  v8bf hi, lo;
#pragma unroll
  for (int j = 0; j < 8; ++j) {
    __bf16 h = (__bf16)v[j];
    hi[j] = h;
    lo[j] = (__bf16)(v[j] - (float)h);
  }
  const size_t base = (i >> 5) * 64 + (i & 31);
  *(v8bf*)(g_w_split + base)      = hi;   // 16B stores, 16B aligned
  *(v8bf*)(g_w_split + base + 32) = lo;
}

// ------------------------------- main kernel --------------------------------
__launch_bounds__(kThreads, 2)
__global__ void moe_gate_kernel(const float* __restrict__ x,      // [T,H]
                                const float* __restrict__ bias,   // [E]
                                float* __restrict__ out)          // [T*8 idx | T*8 wts]
{
  // Transposed score exchange buffer: [expert][token-in-block] = 64 KB.
  // Phase-2 reads (thread == token) are bank-conflict free.
  __shared__ float s_sc[kE][kTokPerBlock];

  const int lane = threadIdx.x & 31;
  const int wave = threadIdx.x >> 5;
  const int grp  = wave >> 1;        // token group (32 tokens each)
  const int half = wave & 1;         // expert half (128 experts each)
  const int tok0 = blockIdx.x * kTokPerBlock;

  const int lm = lane & 15;          // row/col within 16
  const int lh = lane >> 4;          // half-lane (ISA fragment swizzle)

  const int eBase = half * 128;

  // ------------------- GEMM: 32 tokens x 128 experts per wave --------------
  v8f acc[2][8];
#pragma unroll
  for (int s = 0; s < 2; ++s)
#pragma unroll
    for (int nt = 0; nt < 8; ++nt)
      acc[s][nt] = (v8f)0.0f;

  const float* baseA[2] = {
    x + (size_t)(tok0 + grp * 32 + 0 * 16 + lm) * kH,
    x + (size_t)(tok0 + grp * 32 + 1 * 16 + lm) * kH
  };
  // Per-lane fragment base in split W: expert lane-column + K-chunk swizzle.
  // Lane's hi chunk at +0, lo chunk at +32 elements (immediate offsets).
  const __bf16* baseB = g_w_split + (size_t)(eBase + lm) * kH * 2 + lh * 16;

  // A raw-data single buffer: loads for step s+1 are issued right after the
  // step-s conversion reads it (WAR via issue order); the 48 WMMAs of step s
  // cover their latency.
  float4 araw[2][4];
  auto loadA = [&](int step) {
    const int c0 = step * 32 + lh * 16;       // A contiguous K order (matches B)
    const int c1 = step * 32 + lh * 16 + 8;
#pragma unroll
    for (int s = 0; s < 2; ++s) {
      araw[s][0] = *(const float4*)(baseA[s] + c0);
      araw[s][1] = *(const float4*)(baseA[s] + c0 + 4);
      araw[s][2] = *(const float4*)(baseA[s] + c1);
      araw[s][3] = *(const float4*)(baseA[s] + c1 + 4);
    }
  };

  // One K-step: convert A, (optionally) prefetch next A, 8 n-tiles x 6 WMMA
  // with direct B fragment loads (base + immediate offsets).
  auto kstep = [&](int step, bool prefetchA) {
    v16bf a_hi[2], a_lo[2];
#pragma unroll
    for (int s = 0; s < 2; ++s) {
      cvt_split8(araw[s][0], araw[s][1], a_hi[s], a_lo[s], 0);
      cvt_split8(araw[s][2], araw[s][3], a_hi[s], a_lo[s], 8);
    }
    if (prefetchA) loadA(step + 1);

    const __bf16* pB = baseB + (size_t)step * 64;   // 64 elems per K-block row
#pragma unroll
    for (int nt = 0; nt < 8; ++nt) {
      const __bf16* p = pB + (size_t)nt * 16 * kH * 2;
      // b_hi: both uses adjacent (short live range, no 8-wide batching).
      v16bf b_hi = *(const v16bf*)(p);          // 2 x b128
      acc[0][nt] = wmma_bf16(a_hi[0], b_hi, acc[0][nt]);  // hi*hi
      acc[1][nt] = wmma_bf16(a_hi[1], b_hi, acc[1][nt]);
      acc[0][nt] = wmma_bf16(a_lo[0], b_hi, acc[0][nt]);  // lo*hi
      acc[1][nt] = wmma_bf16(a_lo[1], b_hi, acc[1][nt]);
      // b_lo: single pair of uses at the end.
      v16bf b_lo = *(const v16bf*)(p + 32);     // 2 x b128, +64 B
      acc[0][nt] = wmma_bf16(a_hi[0], b_lo, acc[0][nt]);  // hi*lo
      acc[1][nt] = wmma_bf16(a_hi[1], b_lo, acc[1][nt]);
    }
  };

  loadA(0);
  for (int step = 0; step < kSteps - 1; ++step)   // branchless body
    kstep(step, true);
  kstep(kSteps - 1, false);                       // peeled tail

  // ------------------- Epilogue: sigmoid -> LDS (transposed) ---------------
  // C/D layout: VGPR r -> M = lh*8 + r (within 16), N = lm.
#pragma unroll
  for (int s = 0; s < 2; ++s) {
#pragma unroll
    for (int nt = 0; nt < 8; ++nt) {
      const int n  = eBase + nt * 16 + lm;
      const int m0 = grp * 32 + s * 16 + lh * 8;
      float4 f0, f1;
      f0.x = sigmoidf_fast(acc[s][nt][0]);
      f0.y = sigmoidf_fast(acc[s][nt][1]);
      f0.z = sigmoidf_fast(acc[s][nt][2]);
      f0.w = sigmoidf_fast(acc[s][nt][3]);
      f1.x = sigmoidf_fast(acc[s][nt][4]);
      f1.y = sigmoidf_fast(acc[s][nt][5]);
      f1.z = sigmoidf_fast(acc[s][nt][6]);
      f1.w = sigmoidf_fast(acc[s][nt][7]);
      *(float4*)&s_sc[n][m0]     = f0;   // 32B-aligned LDS b128 stores
      *(float4*)&s_sc[n][m0 + 4] = f1;
    }
  }

  __syncthreads();

  // ------------------- Phase 2: per-token 8-smallest of (score + bias) -----
  const int t = threadIdx.x;
  if (t < kTokPerBlock) {
    float bv[kTopK];
    int   bi[kTopK];
#pragma unroll
    for (int j = 0; j < kTopK; ++j) { bv[j] = 1e30f; bi[j] = 0; }

    for (int e = 0; e < kE; ++e) {
      const float v = s_sc[e][t] + bias[e];     // biased score
      if (v < bv[kTopK - 1]) {                  // strict < keeps earlier-index
        bv[kTopK - 1] = v;                      // tie-break (matches top_k)
        bi[kTopK - 1] = e;
#pragma unroll
        for (int q = kTopK - 1; q > 0; --q) {
          if (bv[q] < bv[q - 1]) {
            float tv = bv[q]; bv[q] = bv[q - 1]; bv[q - 1] = tv;
            int   ti = bi[q]; bi[q] = bi[q - 1]; bi[q - 1] = ti;
          }
        }
      }
    }

    // Gather ORIGINAL scores at selected indices, normalize, scale.
    float sel[kTopK];
    float sum = 0.0f;
#pragma unroll
    for (int j = 0; j < kTopK; ++j) {
      sel[j] = s_sc[bi[j]][t];
      sum += sel[j];
    }
    const float inv = kScale / sum;

    const size_t tg = (size_t)tok0 + t;
    float* out_idx = out;                        // first T*8 floats: indices
    float* out_wts = out + (size_t)kT * kTopK;   // next  T*8 floats: weights
#pragma unroll
    for (int j = 0; j < kTopK; ++j) {
      out_idx[tg * kTopK + j] = (float)bi[j];
      out_wts[tg * kTopK + j] = sel[j] * inv;
    }
  }
}

extern "C" void kernel_launch(void* const* d_in, const int* in_sizes, int n_in,
                              void* d_out, int out_size, void* d_ws, size_t ws_size,
                              hipStream_t stream) {
  (void)in_sizes; (void)n_in; (void)out_size; (void)d_ws; (void)ws_size;
  const float* x    = (const float*)d_in[0];   // [T,H] fp32
  const float* w    = (const float*)d_in[1];   // [E,H] fp32
  const float* bias = (const float*)d_in[2];   // [E]   fp32
  float* out = (float*)d_out;                  // [T*8 idx | T*8 wts]

  // Split W once per call (4 MB r/w, ~0.3 us) into the device global.
  const size_t wElems = (size_t)kE * kH;
  hipLaunchKernelGGL(split_w_kernel, dim3((unsigned)(wElems / (256 * 8))),
                     dim3(256), 0, stream, w);

  dim3 grid(kT / kTokPerBlock);   // 256 blocks
  dim3 block(kThreads);           // 128 threads = 4 waves
  hipLaunchKernelGGL(moe_gate_kernel, grid, block, 0, stream, x, bias, out);
}
